// PermutationGenerator_52982716563611
// MI455X (gfx1250) — matrix operand order repaired and checked by reading, not compiled, not run on hardware
//
#include <hip/hip_runtime.h>
#include <hip/hip_bf16.h>

// ---- problem constants (match reference) ----
#define BATCH 8192
#define NN    64
#define DD    128
#define LAT   256
#define TEMP_INV 10.0f           // 1/TEMP, TEMP=0.1, NOISE_FACTOR=1.0
#define NEGV  (-1000000000.0f)
#define SINKHORN_ITERS 20

typedef __attribute__((ext_vector_type(16))) __bf16 v16bf;
typedef __attribute__((ext_vector_type(8)))  float  v8f;

union BF16x16 { unsigned short u[16]; v16bf v; };

#if __has_builtin(__builtin_amdgcn_global_load_async_to_lds_b128) && \
    __has_builtin(__builtin_amdgcn_s_wait_asynccnt)
#define HAVE_ASYNC_LDS 1
// 128-bit payload type exactly as the builtin expects (vector_size int4),
// with explicit global (AS1) / LDS (AS3) qualifications for the pointers.
typedef int v4i __attribute__((vector_size(16)));
typedef __attribute__((address_space(1))) v4i v4i_g;
typedef __attribute__((address_space(3))) v4i v4i_l;
#else
#define HAVE_ASYNC_LDS 0
#endif

__device__ __forceinline__ unsigned short f2bf(float f) {
  unsigned int u = __float_as_uint(f);
  unsigned int r = (u + 0x7fffu + ((u >> 16) & 1u)) >> 16;   // RNE
  return (unsigned short)r;
}

// Pre-transpose + bf16-convert the shared weights once per launch.
// w1t[l][d] = W1[d][l]  (LAT x DD),  w2t[m][l] = W2[l][m]  (NN x LAT)
__global__ void prep_weights_kernel(const float* __restrict__ W1,
                                    const float* __restrict__ W2,
                                    unsigned short* __restrict__ w1t,
                                    unsigned short* __restrict__ w2t) {
  int t = blockIdx.x * 256 + threadIdx.x;
  if (t < LAT * DD) {
    int l = t / DD, d = t % DD;
    w1t[t] = f2bf(W1[d * LAT + l]);
  }
  int t2 = t - LAT * DD;
  if (t2 >= 0 && t2 < NN * LAT) {
    int m = t2 / LAT, l = t2 % LAT;
    w2t[t2] = f2bf(W2[l * NN + m]);
  }
}

// One workgroup (8 wave32) per batch element; fully fused pipeline.
__global__ __launch_bounds__(256) void perm_gen_kernel(
    const float* __restrict__ X,        // [B][NN][DD]
    const int*   __restrict__ set_size, // [B]
    const float* __restrict__ GN,       // [B][NN][NN] gumbel noise
    const float* __restrict__ b1,       // [LAT]
    const float* __restrict__ b2,       // [NN]
    const unsigned short* __restrict__ w1t, // [LAT][DD] bf16
    const unsigned short* __restrict__ w2t, // [NN][LAT] bf16
    float* __restrict__ out_set,        // [B]
    float* __restrict__ out_perm)       // [B][NN][DD]
{
  __shared__ unsigned short xb[NN * DD];                       // 16 KB bf16 [i][d]
  __shared__ unsigned short hb[NN * LAT];                      // 32 KB bf16 [i][l]
  __shared__ __attribute__((aligned(16))) float la[NN * NN];   // 16 KB f32

  const int b    = blockIdx.x;
  const int t    = threadIdx.x;
  const int w    = t >> 5;        // wave id 0..7
  const int lane = t & 31;
  const int half = lane >> 4;     // 0: lanes 0-15, 1: lanes 16-31
  const int lm   = lane & 15;
  const int k    = set_size[b];

  if (t == 0) out_set[b] = (float)k;

  // ---- kick off async DMA of gumbel noise into the la buffer (idle until
  //      GEMM2 epilogue); overlaps with x staging + all of GEMM1 ----
  const float* gn = GN + (size_t)b * (NN * NN);
#if HAVE_ASYNC_LDS
  {
    #pragma unroll
    for (int c = 0; c < 4; ++c) {
      const int idx = (c * 256 + t) * 4;   // float index, 16B granules
      __builtin_amdgcn_global_load_async_to_lds_b128(
          (v4i_g*)(gn + idx), (v4i_l*)(&la[idx]), 0, 0);
    }
  }
#else
  {
    const float4* g4 = (const float4*)gn;
    float4* l4 = (float4*)la;
    for (int i = t; i < (NN * NN) / 4; i += 256) l4[i] = g4[i];
  }
#endif

  // ---- stage x tile into LDS as bf16 (coalesced f32 reads) ----
  const float* xg = X + (size_t)b * (NN * DD);
  for (int idx = t; idx < NN * DD; idx += 256) xb[idx] = f2bf(xg[idx]);
  __syncthreads();

  // ---- GEMM1: h = relu(x @ W1 + b1), 64x128 * 128x256 ----
  for (int nn = 0; nn < 2; ++nn) {
    const int nt  = w * 2 + nn;          // N-tile of 256
    const int col = nt * 16 + lm;        // latent column for this lane
    for (int mt = 0; mt < 4; ++mt) {
      const int row = mt * 16 + lm;      // A row (M = lane%16)
      v8f acc = {};
      for (int kt = 0; kt < 4; ++kt) {
        const int k0 = kt * 32;
        BF16x16 A, Bm;
        const unsigned short* ar = &xb[row * DD + k0 + 16 * half];
        #pragma unroll
        for (int e = 0; e < 16; ++e) A.u[e] = ar[e];
        const unsigned short* br = &w1t[col * DD + k0 + 16 * half];
        #pragma unroll
        for (int e = 0; e < 16; ++e) Bm.u[e] = br[e];
        acc = __builtin_amdgcn_wmma_f32_16x16x32_bf16(
            false, A.v, false, Bm.v, (short)0, acc, false, false);
      }
      const float bias = b1[col];
      #pragma unroll
      for (int v = 0; v < 8; ++v) {
        float h = acc[v] + bias;
        h = h > 0.f ? h : 0.f;
        const int r = mt * 16 + v + 8 * half;   // C/D row layout
        hb[r * LAT + col] = f2bf(h);
      }
    }
  }
#if HAVE_ASYNC_LDS
  __builtin_amdgcn_s_wait_asynccnt(0);   // gumbel DMA visible in LDS
#endif
  __syncthreads();

  // ---- GEMM2: net = h @ W2 + b2; log_alpha = (net + gumbel)/TEMP ----
  // gumbel is read from LDS (la) and overwritten in place by the same thread.
  {
    for (int i = 0; i < 2; ++i) {
      const int id  = w * 2 + i;           // 16 tiles over 8 waves
      const int mt  = id >> 2, nt = id & 3;
      const int row = mt * 16 + lm;
      const int col = nt * 16 + lm;
      v8f acc = {};
      for (int kt = 0; kt < 8; ++kt) {
        const int k0 = kt * 32;
        BF16x16 A, Bm;
        const unsigned short* ar = &hb[row * LAT + k0 + 16 * half];
        #pragma unroll
        for (int e = 0; e < 16; ++e) A.u[e] = ar[e];
        const unsigned short* br = &w2t[col * LAT + k0 + 16 * half];
        #pragma unroll
        for (int e = 0; e < 16; ++e) Bm.u[e] = br[e];
        acc = __builtin_amdgcn_wmma_f32_16x16x32_bf16(
            false, A.v, false, Bm.v, (short)0, acc, false, false);
      }
      const float bias = b2[col];
      #pragma unroll
      for (int v = 0; v < 8; ++v) {
        const int r = mt * 16 + v + 8 * half;
        const float noise = la[r * NN + col];          // DMA'd gumbel
        la[r * NN + col] = (acc[v] + bias + noise) * TEMP_INV;
      }
    }
  }
  __syncthreads();

  // ---- Sinkhorn: 20 iters, log-domain, mask derived from set_size ----
  // 4 threads per row (then per column); 64-wide logsumexp via shfl_xor.
  {
    const int rr  = t >> 2;          // row (then column) index 0..63
    const int seg = (t & 3) * 16;    // 16-element segment owned by this thread
    for (int it = 0; it < SINKHORN_ITERS; ++it) {
      // row normalization (reduce over columns)
      {
        float vb[16]; float m = NEGV;
        #pragma unroll
        for (int j = 0; j < 16; ++j) {
          const int c = seg + j;
          const float v = (rr < k && c < k) ? la[rr * NN + c] : NEGV;
          vb[j] = v; m = fmaxf(m, v);
        }
        m = fmaxf(m, __shfl_xor(m, 1)); m = fmaxf(m, __shfl_xor(m, 2));
        float s = 0.f;
        #pragma unroll
        for (int j = 0; j < 16; ++j) s += __expf(vb[j] - m);
        s += __shfl_xor(s, 1); s += __shfl_xor(s, 2);
        const float lse = m + __logf(s);
        #pragma unroll
        for (int j = 0; j < 16; ++j) {
          const int c = seg + j;
          la[rr * NN + c] = (rr < k && c < k) ? (vb[j] - lse) : NEGV;
        }
      }
      __syncthreads();
      // column normalization (reduce over rows)
      {
        float vb[16]; float m = NEGV;
        #pragma unroll
        for (int j = 0; j < 16; ++j) {
          const int i = seg + j;
          const float v = (i < k && rr < k) ? la[i * NN + rr] : NEGV;
          vb[j] = v; m = fmaxf(m, v);
        }
        m = fmaxf(m, __shfl_xor(m, 1)); m = fmaxf(m, __shfl_xor(m, 2));
        float s = 0.f;
        #pragma unroll
        for (int j = 0; j < 16; ++j) s += __expf(vb[j] - m);
        s += __shfl_xor(s, 1); s += __shfl_xor(s, 2);
        const float lse = m + __logf(s);
        #pragma unroll
        for (int j = 0; j < 16; ++j) {
          const int i = seg + j;
          la[i * NN + rr] = (i < k && rr < k) ? (vb[j] - lse) : NEGV;
        }
      }
      __syncthreads();
    }
  }

  // ---- GEMM3: permuted[j][f] = sum_i exp(la[i][j]) * x[i][f] ----
  // A[M=j][K=i] = post^T, generated on the fly (exp + bf16 convert).
  {
    float* og = out_perm + (size_t)b * (NN * DD);
    const int nt   = w;                 // f-tile 0..7
    const int fcol = nt * 16 + lm;      // B column (f)
    for (int mt = 0; mt < 4; ++mt) {
      const int jcol = mt * 16 + lm;    // A row (j)
      v8f acc = {};
      #pragma unroll
      for (int kt = 0; kt < 2; ++kt) {
        const int k0 = kt * 32;
        BF16x16 A, Bm;
        #pragma unroll
        for (int e = 0; e < 8; ++e) {
          const int i1 = k0 + 16 * half + e;
          const int i2 = k0 + 16 * half + 8 + e;
          A.u[e]     = f2bf(__expf(la[i1 * NN + jcol]));
          A.u[8 + e] = f2bf(__expf(la[i2 * NN + jcol]));
        }
        #pragma unroll
        for (int e = 0; e < 16; ++e) {
          const int i = k0 + 16 * half + e;
          Bm.u[e] = xb[i * DD + fcol];
        }
        acc = __builtin_amdgcn_wmma_f32_16x16x32_bf16(
            false, A.v, false, Bm.v, (short)0, acc, false, false);
      }
      #pragma unroll
      for (int v = 0; v < 8; ++v) {
        const int j = mt * 16 + v + 8 * half;
        og[j * DD + fcol] = acc[v];     // coalesced 64B stores per lane group
      }
    }
  }
}

extern "C" void kernel_launch(void* const* d_in, const int* in_sizes, int n_in,
                              void* d_out, int out_size, void* d_ws, size_t ws_size,
                              hipStream_t stream) {
  (void)in_sizes; (void)n_in; (void)out_size; (void)ws_size;
  const float* X  = (const float*)d_in[0];
  const int*   ss = (const int*)d_in[1];
  // d_in[2] = maskB: fully determined by set_size, ignored.
  const float* GN = (const float*)d_in[3];
  const float* W1 = (const float*)d_in[4];
  const float* b1 = (const float*)d_in[5];
  const float* W2 = (const float*)d_in[6];
  const float* b2 = (const float*)d_in[7];

  unsigned short* w1t = (unsigned short*)d_ws;      // LAT*DD bf16 = 64 KB
  unsigned short* w2t = w1t + LAT * DD;             // NN*LAT bf16 = 32 KB

  float* out_set  = (float*)d_out;                  // [B]
  float* out_perm = out_set + BATCH;                // [B][NN][DD]

  const int prep_elems = LAT * DD + NN * LAT;
  prep_weights_kernel<<<(prep_elems + 255) / 256, 256, 0, stream>>>(W1, W2, w1t, w2t);
  perm_gen_kernel<<<BATCH, 256, 0, stream>>>(X, ss, GN, b1, b2, w1t, w2t,
                                             out_set, out_perm);
}
